// MultiHeadAttention_51402168599031
// MI455X (gfx1250) — matrix operand order repaired and checked by reading, not compiled, not run on hardware
//
#include <hip/hip_runtime.h>

typedef __attribute__((ext_vector_type(16))) __bf16 v16bf;
typedef __attribute__((ext_vector_type(8)))  float  v8f;
typedef unsigned short u16;

union Frag { uint4 u[2]; v16bf v; };

__device__ __forceinline__ u16 f2bf(float f) {
  unsigned int u = __builtin_bit_cast(unsigned int, f);
  u += 0x7FFFu + ((u >> 16) & 1u);          // round-to-nearest-even
  return (u16)(u >> 16);
}

// ---------------------------------------------------------------------------
// fp32 -> bf16 conversion
// ---------------------------------------------------------------------------
__global__ void cvt_f32_bf16(const float* __restrict__ in, u16* __restrict__ out, int n) {
  int i = blockIdx.x * blockDim.x + threadIdx.x;
  if (i < n) out[i] = f2bf(in[i]);
}

__device__ __forceinline__ void store_out(float* __restrict__ outF, u16* __restrict__ outB,
                                          int mode, int N, int S, int HN,
                                          int m, int n, float val) {
  if (mode == 0) {
    outF[(size_t)m * N + n] = val;
  } else {
    const int b = m / S, s = m - b * S;
    const int h = n >> 6, d = n & 63;
    size_t idx = (mode == 1) ? (((size_t)(b * HN + h) * S + s) * 64 + d)
                             : (((size_t)(b * HN + h) * 64 + d) * S + s);
    outB[idx] = f2bf(val);
  }
}

// ---------------------------------------------------------------------------
// out[m][n] = sum_k A[m][k] * W[n][k]   (i.e. A @ W^T), bf16 in, WMMA bf16.
// mode 0: f32 flat [M][N] -> outF
// mode 1: bf16 head layout  [B][H][S][64]   -> outB   (Q/K projections)
// mode 2: bf16 head layout^T [B][H][64][S]  -> outB   (V projection)
// Block: 256 threads = 8 waves (2x4), wave tile 32x32 (4 accums),
// block tile 64(M) x 128(N).  Per 32-K step: 8 b128 loads -> 4 WMMAs.
// ---------------------------------------------------------------------------
__global__ __launch_bounds__(256)
void gemm_bf16(const u16* __restrict__ A, const u16* __restrict__ W,
               float* __restrict__ outF, u16* __restrict__ outB,
               int M, int N, int K, int S, int mode)
{
  const int tid  = threadIdx.x;
  const int wave = tid >> 5;
  const int lane = tid & 31;
  const int l16  = lane & 15;
  const int half = lane >> 4;
  const int wm = wave >> 2;              // 0..1
  const int wn = wave & 3;               // 0..3
  const int mBase = blockIdx.x * 64 + wm * 32;
  const int nBase = blockIdx.y * 128 + wn * 32;

  const u16* arow0 = A + (size_t)(mBase + l16) * K;
  const u16* arow1 = A + (size_t)(mBase + 16 + l16) * K;
  const u16* brow0 = W + (size_t)(nBase + l16) * K;
  const u16* brow1 = W + (size_t)(nBase + 16 + l16) * K;

  v8f c00 = {}, c01 = {}, c10 = {}, c11 = {};
  for (int k0 = 0; k0 < K; k0 += 32) {
    Frag a0, a1, b0, b1;
    // A fragment: lanes 0-15 rows, K 0..7 & 16..23 ; lanes 16-31: K 8..15 & 24..31
    a0.u[0] = *(const uint4*)(arow0 + k0 + half * 8);
    a0.u[1] = *(const uint4*)(arow0 + k0 + 16 + half * 8);
    a1.u[0] = *(const uint4*)(arow1 + k0 + half * 8);
    a1.u[1] = *(const uint4*)(arow1 + k0 + 16 + half * 8);
    // B fragment: lane = column n, lanes 0-15 hold K 0..7 & 16..23, lanes 16-31 K 8..15 & 24..31
    b0.u[0] = *(const uint4*)(brow0 + k0 + half * 8);
    b0.u[1] = *(const uint4*)(brow0 + k0 + 16 + half * 8);
    b1.u[0] = *(const uint4*)(brow1 + k0 + half * 8);
    b1.u[1] = *(const uint4*)(brow1 + k0 + 16 + half * 8);
    __builtin_prefetch(arow0 + k0 + 512, 0, 3);   // WGP-scope prefetch, 512 elements (16 K-steps) ahead
    __builtin_prefetch(brow0 + k0 + 512, 0, 3);
    c00 = __builtin_amdgcn_wmma_f32_16x16x32_bf16(false, a0.v, false, b0.v, (short)0, c00, false, false);
    c01 = __builtin_amdgcn_wmma_f32_16x16x32_bf16(false, a0.v, false, b1.v, (short)0, c01, false, false);
    c10 = __builtin_amdgcn_wmma_f32_16x16x32_bf16(false, a1.v, false, b0.v, (short)0, c10, false, false);
    c11 = __builtin_amdgcn_wmma_f32_16x16x32_bf16(false, a1.v, false, b1.v, (short)0, c11, false, false);
  }

  const int HN = N >> 6;   // heads
  #pragma unroll
  for (int r = 0; r < 8; ++r) {
    const int m0 = mBase + r + half * 8;        // C layout: vgpr r, +8 for lane half
    const int m1 = m0 + 16;
    const int n0 = nBase + l16;
    const int n1 = n0 + 16;
    store_out(outF, outB, mode, N, S, HN, m0, n0, c00[r]);
    store_out(outF, outB, mode, N, S, HN, m0, n1, c01[r]);
    store_out(outF, outB, mode, N, S, HN, m1, n0, c10[r]);
    store_out(outF, outB, mode, N, S, HN, m1, n1, c11[r]);
  }
}

// ---------------------------------------------------------------------------
// Flash attention, causal. One block per (64-row q tile, b*h).
// Qh/Kh: [B*H][S][64] bf16.  Vt: [B*H][64][S] bf16.  Ctx: [B][S][H*64] bf16.
// 8 waves cover the 4x4 grid of 16x16 tiles of the 64x64 score block
// (wave w -> row tile mi=w>>1, col tiles ni0=(w&1)*2, +1).
// ---------------------------------------------------------------------------
__global__ __launch_bounds__(256)
void attn_fused(const u16* __restrict__ Qh, const u16* __restrict__ Kh,
                const u16* __restrict__ Vt, u16* __restrict__ Ctx,
                int S, int H, int D, float scale)
{
  __shared__ __align__(16) float sS[64 * 72];    // scores, f32 (stride 288B = 18*16B)
  __shared__ __align__(16) u16   pSh[64 * 72];   // P, bf16   (stride 144B = 9*16B)
  __shared__ float rowScale[64];

  const int qt = blockIdx.x;
  const int bh = blockIdx.y;
  const int b  = bh / H;
  const int h  = bh - b * H;

  const int tid  = threadIdx.x;
  const int wave = tid >> 5;
  const int lane = tid & 31;
  const int l16  = lane & 15;
  const int half = lane >> 4;
  const int mi  = wave >> 1;
  const int ni0 = (wave & 1) * 2;

  const size_t bhQK = (size_t)bh * S * 64;
  const size_t bhV  = (size_t)bh * 64 * S;

  // Q A-fragments, kept in registers for the whole kt loop
  const int qRow = qt * 64 + mi * 16 + l16;
  const u16* qbase = Qh + bhQK + (size_t)qRow * 64;
  Frag qa[2];
  #pragma unroll
  for (int j = 0; j < 2; ++j) {
    qa[j].u[0] = *(const uint4*)(qbase + j * 32 + half * 8);
    qa[j].u[1] = *(const uint4*)(qbase + j * 32 + 16 + half * 8);
  }

  const v8f vzero = {};
  v8f oacc[2];
  oacc[0] = vzero; oacc[1] = vzero;
  float mloc = -1e30f, lloc = 0.0f;

  for (int kt = 0; kt <= qt; ++kt) {
    // ---- scores: S = scale * Q K^T, causal mask ----
    #pragma unroll
    for (int t = 0; t < 2; ++t) {
      const int ni = ni0 + t;
      const int kcol = kt * 64 + ni * 16 + l16;
      const u16* kbase = Kh + bhQK + (size_t)kcol * 64;
      Frag kb0, kb1;
      kb0.u[0] = *(const uint4*)(kbase + half * 8);
      kb0.u[1] = *(const uint4*)(kbase + 16 + half * 8);
      kb1.u[0] = *(const uint4*)(kbase + 32 + half * 8);
      kb1.u[1] = *(const uint4*)(kbase + 32 + 16 + half * 8);
      v8f sc = {};
      sc = __builtin_amdgcn_wmma_f32_16x16x32_bf16(false, qa[0].v, false, kb0.v, (short)0, sc, false, false);
      sc = __builtin_amdgcn_wmma_f32_16x16x32_bf16(false, qa[1].v, false, kb1.v, (short)0, sc, false, false);
      #pragma unroll
      for (int r = 0; r < 8; ++r) {
        const int lrow = mi * 16 + r + half * 8;
        const int grow = qt * 64 + lrow;
        const int gcol = kt * 64 + ni * 16 + l16;
        float v = sc[r] * scale;
        if (gcol > grow) v = -1e30f;
        sS[lrow * 72 + ni * 16 + l16] = v;
      }
    }
    __syncthreads();

    // ---- online softmax: one thread per row, float4 LDS reads ----
    if (tid < 64) {
      const float4* srow4 = (const float4*)&sS[tid * 72];
      float mx = mloc;
      #pragma unroll
      for (int j = 0; j < 16; ++j) {
        float4 s4 = srow4[j];
        mx = fmaxf(mx, fmaxf(fmaxf(s4.x, s4.y), fmaxf(s4.z, s4.w)));
      }
      const float rs = __expf(mloc - mx);
      float sum = 0.0f;
      u16* prow = &pSh[tid * 72];
      #pragma unroll
      for (int j = 0; j < 16; ++j) {
        float4 s4 = srow4[j];
        float p0 = __expf(s4.x - mx), p1 = __expf(s4.y - mx);
        float p2 = __expf(s4.z - mx), p3 = __expf(s4.w - mx);
        sum += (p0 + p1) + (p2 + p3);
        prow[j * 4 + 0] = f2bf(p0);
        prow[j * 4 + 1] = f2bf(p1);
        prow[j * 4 + 2] = f2bf(p2);
        prow[j * 4 + 3] = f2bf(p3);
      }
      lloc = lloc * rs + sum;
      mloc = mx;
      rowScale[tid] = rs;
    }
    __syncthreads();

    // ---- rescale O, then O += P @ V ----
    float rsr[8];
    #pragma unroll
    for (int r = 0; r < 8; ++r) rsr[r] = rowScale[mi * 16 + r + half * 8];
    #pragma unroll
    for (int t = 0; t < 2; ++t)
      #pragma unroll
      for (int r = 0; r < 8; ++r) oacc[t][r] *= rsr[r];

    Frag pa[2];
    const u16* prow = &pSh[(mi * 16 + l16) * 72];
    #pragma unroll
    for (int j = 0; j < 2; ++j) {
      pa[j].u[0] = *(const uint4*)(prow + j * 32 + half * 8);
      pa[j].u[1] = *(const uint4*)(prow + j * 32 + 16 + half * 8);
    }
    #pragma unroll
    for (int t = 0; t < 2; ++t) {
      const int d = (ni0 + t) * 16 + l16;
      const u16* vbase = Vt + bhV + (size_t)d * S + kt * 64;
      Frag vb0, vb1;
      vb0.u[0] = *(const uint4*)(vbase + half * 8);
      vb0.u[1] = *(const uint4*)(vbase + 16 + half * 8);
      vb1.u[0] = *(const uint4*)(vbase + 32 + half * 8);
      vb1.u[1] = *(const uint4*)(vbase + 32 + 16 + half * 8);
      oacc[t] = __builtin_amdgcn_wmma_f32_16x16x32_bf16(false, pa[0].v, false, vb0.v, (short)0, oacc[t], false, false);
      oacc[t] = __builtin_amdgcn_wmma_f32_16x16x32_bf16(false, pa[1].v, false, vb1.v, (short)0, oacc[t], false, false);
    }
    __syncthreads();
  }

  if (tid < 64) rowScale[tid] = (lloc > 0.0f) ? (1.0f / lloc) : 0.0f;
  __syncthreads();

  #pragma unroll
  for (int r = 0; r < 8; ++r) {
    const int lrow = mi * 16 + r + half * 8;
    const float inv = rowScale[lrow];
    const int sgl = qt * 64 + lrow;
    #pragma unroll
    for (int t = 0; t < 2; ++t) {
      const int d = (ni0 + t) * 16 + l16;
      Ctx[(size_t)(b * S + sgl) * D + h * 64 + d] = f2bf(oacc[t][r] * inv);
    }
  }
}

// ---------------------------------------------------------------------------
// Host launcher
// ---------------------------------------------------------------------------
extern "C" void kernel_launch(void* const* d_in, const int* in_sizes, int n_in,
                              void* d_out, int out_size, void* d_ws, size_t ws_size,
                              hipStream_t stream) {
  const float* q  = (const float*)d_in[0];
  const float* k  = (const float*)d_in[1];
  const float* v  = (const float*)d_in[2];
  // d_in[3] = causal mask (handled analytically)
  const float* Wq = (const float*)d_in[4];
  const float* Wk = (const float*)d_in[5];
  const float* Wv = (const float*)d_in[6];
  const float* Wo = (const float*)d_in[7];

  const int D = 1024, S = 2048, H = 16;
  const int BS = in_sizes[0] / D;       // B*S
  const int B  = BS / S;
  const float scale = 0.125f;           // 1/sqrt(64)

  char* ws = (char*)d_ws;
  size_t off = 0;
  auto carve = [&](size_t bytes) -> u16* {
    u16* p = (u16*)(ws + off);
    off += (bytes + 255) & ~(size_t)255;
    return p;
  };
  const size_t szX = (size_t)BS * D * sizeof(u16);
  const size_t szW = (size_t)D * D * sizeof(u16);
  u16* qb  = carve(szX);
  u16* kb  = carve(szX);
  u16* vb  = carve(szX);
  u16* Wqb = carve(szW);
  u16* Wkb = carve(szW);
  u16* Wvb = carve(szW);
  u16* Wob = carve(szW);
  u16* Qh  = carve(szX);
  u16* Kh  = carve(szX);
  u16* Vt  = carve(szX);
  u16* Ctx = carve(szX);

  const int nX = BS * D, nW = D * D;
  cvt_f32_bf16<<<(nX + 255) / 256, 256, 0, stream>>>(q,  qb,  nX);
  cvt_f32_bf16<<<(nX + 255) / 256, 256, 0, stream>>>(k,  kb,  nX);
  cvt_f32_bf16<<<(nX + 255) / 256, 256, 0, stream>>>(v,  vb,  nX);
  cvt_f32_bf16<<<(nW + 255) / 256, 256, 0, stream>>>(Wq, Wqb, nW);
  cvt_f32_bf16<<<(nW + 255) / 256, 256, 0, stream>>>(Wk, Wkb, nW);
  cvt_f32_bf16<<<(nW + 255) / 256, 256, 0, stream>>>(Wv, Wvb, nW);
  cvt_f32_bf16<<<(nW + 255) / 256, 256, 0, stream>>>(Wo, Wob, nW);

  dim3 gg(BS / 64, D / 128);
  gemm_bf16<<<gg, 256, 0, stream>>>(qb, Wqb, nullptr, Qh, BS, D, D, S, 1);
  gemm_bf16<<<gg, 256, 0, stream>>>(kb, Wkb, nullptr, Kh, BS, D, D, S, 1);
  gemm_bf16<<<gg, 256, 0, stream>>>(vb, Wvb, nullptr, Vt, BS, D, D, S, 2);

  attn_fused<<<dim3(S / 64, B * H), 256, 0, stream>>>(Qh, Kh, Vt, Ctx, S, H, D, scale);

  gemm_bf16<<<gg, 256, 0, stream>>>(Ctx, Wob, (float*)d_out, nullptr, BS, D, D, S, 0);
}
